// TimeAwareGQA_87402584473692
// MI455X (gfx1250) — matrix operand order repaired and checked by reading, not compiled, not run on hardware
//
#include <hip/hip_runtime.h>
#include <math.h>

typedef __attribute__((ext_vector_type(16))) _Float16 v16h;
typedef __attribute__((ext_vector_type(8)))  _Float16 v8h;
typedef __attribute__((ext_vector_type(2)))  _Float16 v2h;
typedef __attribute__((ext_vector_type(2)))  __fp16   v2fp;
typedef __attribute__((ext_vector_type(8)))  float    v8f;

#define EMBED   1024
#define KVDIM   256
#define NHEAD   16
#define HDIM    64
#define SEQ     2048
#define BATCH   4
#define MROWS   (BATCH * SEQ)          // 8192
#define QKSCALE 0.125f                 // 1/sqrt(64)

union H16 { v16h v; v8h q[2]; };
union H8  { v8h  v; v2h p[4]; };

// v_cvt_pk_rtz_f16_f32 with a type shim (__fp16 vec2 -> _Float16 vec2)
__device__ __forceinline__ v2h cvt_pk(float x, float y) {
  union { v2fp a; v2h b; } u;
  u.a = __builtin_amdgcn_cvt_pkrtz(x, y);
  return u.b;
}

// 8 halves at p0 + 8 halves at p1 as two 16B vector loads -> fragment register pair
__device__ __forceinline__ v16h load_frag16(const _Float16* p0, const _Float16* p1) {
  H16 u;
  u.q[0] = *(const v8h*)p0;
  u.q[1] = *(const v8h*)p1;
  return u.v;
}

__device__ __forceinline__ v8h pack8(float4 a, float4 b) {
  H8 u;
  u.p[0] = cvt_pk(a.x, a.y);
  u.p[1] = cvt_pk(a.z, a.w);
  u.p[2] = cvt_pk(b.x, b.y);
  u.p[3] = cvt_pk(b.z, b.w);
  return u.v;
}

// ---------------------------------------------------------------------------
// elementwise f32 -> f16 (packed converts, b128 in / b128 out)
// ---------------------------------------------------------------------------
__global__ __launch_bounds__(256) void cvt_f16_kernel(
    const float* __restrict__ in, _Float16* __restrict__ out, int n8) {
  int i = blockIdx.x * 256 + threadIdx.x;
  if (i < n8) {
    const float4* s = (const float4*)(in + (size_t)i * 8);
    float4 a = s[0], b = s[1];
    *(v8h*)(out + (size_t)i * 8) = pack8(a, b);
  }
}

// ---------------------------------------------------------------------------
// W[R,C] f32 row-major  ->  Wt[C,R] f16 row-major (LDS-tiled transpose)
// ---------------------------------------------------------------------------
__global__ __launch_bounds__(256) void transpose_cvt_kernel(
    const float* __restrict__ in, _Float16* __restrict__ out, int R, int C) {
  __shared__ _Float16 tile[32][33];
  const int tx = threadIdx.x & 31;
  const int ty = threadIdx.x >> 5;            // 32x8 threads
  const int r0 = blockIdx.y * 32, c0 = blockIdx.x * 32;
#pragma unroll
  for (int i = ty; i < 32; i += 8)
    tile[i][tx] = (_Float16)in[(size_t)(r0 + i) * C + c0 + tx];
  __syncthreads();
#pragma unroll
  for (int i = ty; i < 32; i += 8)
    out[(size_t)(c0 + i) * R + r0 + tx] = tile[tx][i];
}

// ---------------------------------------------------------------------------
// cumsum of session_deltas (tiny: 4 rows)
// ---------------------------------------------------------------------------
__global__ void cumsum_kernel(const float* __restrict__ deltas,
                              float* __restrict__ ct) {
  int b = threadIdx.x;
  if (b < BATCH) {
    float acc = 0.f;
    for (int s = 0; s < SEQ; ++s) { acc += deltas[b * SEQ + s]; ct[b * SEQ + s] = acc; }
  }
}

// ---------------------------------------------------------------------------
// C = A[M,K](f16, row-major) * Bt[N,K](f16, pre-transposed)   -- no LDS.
// Block = 256 (8 waves); tile 128x64; wave w owns rows [16w,16w+16).
// Fragments are two 16B runs per lane (k 8h..8h+7, 16+8h..16+8h+7) -> global b128 loads.
// MODE 0: C f32 [m][n]   MODE 1: C f16 [m][n]   MODE 2: C f16 transposed [n][m]
// ---------------------------------------------------------------------------
template <int MODE>
__global__ __launch_bounds__(256) void wmma_gemm_f16(
    const _Float16* __restrict__ A, const _Float16* __restrict__ Bt,
    void* __restrict__ Cout, int M, int N, int K) {
  const int tid = threadIdx.x;
  const int wave = tid >> 5, lane = tid & 31;
  const int lhi = lane >> 4, llo = lane & 15;
  const int kb8 = lhi * 8, kb16 = lhi * 16;
  const int m0 = blockIdx.y * 128;
  const int n0 = blockIdx.x * 64;

  const _Float16* arow  = A  + (size_t)(m0 + wave * 16 + llo) * K;
  const _Float16* brow0 = Bt + (size_t)(n0 + llo) * K + kb8;

  v8f acc[4] = {};
  for (int k0 = 0; k0 < K; k0 += 32) {
    if (k0 + 32 < K) __builtin_prefetch(arow + k0 + 32, 0, 1);
    v16h af = load_frag16(arow + k0 + kb8, arow + k0 + 16 + kb8);
#pragma unroll
    for (int j = 0; j < 4; ++j) {
      const _Float16* br = brow0 + (size_t)j * 16 * K + k0;
      v16h bf = load_frag16(br, br + 16);
      acc[j] = __builtin_amdgcn_wmma_f32_16x16x32_f16(
          false, af, false, bf, (short)0, acc[j], false, false);
    }
  }

  const int mr = m0 + wave * 16 + lhi * 8;   // C-layout row base for this lane
  if (MODE == 0) {
    float* C = (float*)Cout;
#pragma unroll
    for (int j = 0; j < 4; ++j)
#pragma unroll
      for (int r = 0; r < 8; ++r)
        C[(size_t)(mr + r) * N + n0 + j * 16 + llo] = acc[j][r];
  } else if (MODE == 1) {
    _Float16* C = (_Float16*)Cout;
#pragma unroll
    for (int j = 0; j < 4; ++j)
#pragma unroll
      for (int r = 0; r < 8; ++r)
        C[(size_t)(mr + r) * N + n0 + j * 16 + llo] = (_Float16)acc[j][r];
  } else {
    // transposed f16 store: lane holds 8 consecutive m -> one b128 per subtile
    _Float16* C = (_Float16*)Cout;   // [N][M]
#pragma unroll
    for (int j = 0; j < 4; ++j) {
      H8 u;
#pragma unroll
      for (int r = 0; r < 4; ++r)
        u.p[r] = cvt_pk(acc[j][2 * r], acc[j][2 * r + 1]);
      *(v8h*)(C + (size_t)(n0 + j * 16 + llo) * M + mr) = u.v;
    }
  }
}

// ---------------------------------------------------------------------------
// Fused flash GQA with time bias. Block = 128 (4 waves); grid (S/64, B*H).
// q16[m][E], k16[m][KV] row-major f16; vT[KV][M] transposed f16.
// No block barriers: only wave-private LDS for the P (C->A layout) round-trip.
// ---------------------------------------------------------------------------
#define P_LD 40   // padded row stride (halves) of P tile: 80B rows, 16B aligned

__global__ __launch_bounds__(128) void flash_gqa_kernel(
    const _Float16* __restrict__ qp, const _Float16* __restrict__ kp,
    const _Float16* __restrict__ vT, const float* __restrict__ ct,
    const float* __restrict__ mask, const float* __restrict__ tbw_ptr,
    _Float16* __restrict__ out) {
  __shared__ _Float16 Ps[4][16 * P_LD];

  const int tid = threadIdx.x;
  const int wave = tid >> 5, lane = tid & 31;
  const int lhi = lane >> 4, llo = lane & 15;
  const int kb8 = lhi * 8, kb16 = lhi * 16;

  const int q0 = blockIdx.x * 64;
  const int bh = blockIdx.y;
  const int b = bh >> 4;             // H = 16
  const int h = bh & 15;
  const int g = h >> 2;              // n_rep = 4
  const float tbw = tbw_ptr[0];

  // Q fragments (D=64 -> two 32-deep K-steps), register-resident
  const int qrow = q0 + wave * 16 + llo;
  const _Float16* qr = qp + (size_t)(b * SEQ + qrow) * EMBED + h * HDIM;
  v16h qf[2];
#pragma unroll
  for (int ks = 0; ks < 2; ++ks)
    qf[ks] = load_frag16(qr + ks * 32 + kb8, qr + ks * 32 + 16 + kb8);

  float ctq[8];
#pragma unroll
  for (int r = 0; r < 8; ++r)
    ctq[r] = ct[b * SEQ + q0 + wave * 16 + r + lhi * 8];

  v8f oacc[4] = {};
  float mrun[8], lrun[8];
#pragma unroll
  for (int r = 0; r < 8; ++r) { mrun[r] = -1e30f; lrun[r] = 0.f; }

  _Float16* pw = &Ps[wave][0];

  for (int kt = 0; kt < SEQ; kt += 64) {
    // ---- S = Q K^T : K fragments contiguous from k16 rows ----
    v8f sc[4] = {};
#pragma unroll
    for (int j = 0; j < 4; ++j) {
      const int key = kt + j * 16 + llo;
      const _Float16* kr = kp + (size_t)(b * SEQ + key) * KVDIM + g * HDIM + kb8;
#pragma unroll
      for (int ks = 0; ks < 2; ++ks) {
        v16h bf = load_frag16(kr + ks * 32, kr + ks * 32 + 16);
        sc[j] = __builtin_amdgcn_wmma_f32_16x16x32_f16(
            false, qf[ks], false, bf, (short)0, sc[j], false, false);
      }
    }

    // ---- time bias + mask + online softmax (rows in C layout) ----
    float ctk[4], mk[4];
#pragma unroll
    for (int j = 0; j < 4; ++j) {
      const int key = kt + j * 16 + llo;
      ctk[j] = ct[b * SEQ + key];
      mk[j]  = mask[b * SEQ + key];
    }
#pragma unroll
    for (int r = 0; r < 8; ++r) {
      float mx = -1e30f;
#pragma unroll
      for (int j = 0; j < 4; ++j) {
        float s = sc[j][r] * QKSCALE
                + tbw * log1pf(fabsf(ctq[r] - ctk[j]))
                + (1.0f - mk[j]) * -1e9f;
        sc[j][r] = s;
        mx = fmaxf(mx, s);
      }
#pragma unroll
      for (int off = 8; off >= 1; off >>= 1)
        mx = fmaxf(mx, __shfl_xor(mx, off, 16));
      const float mnew  = fmaxf(mrun[r], mx);
      const float alpha = __expf(mrun[r] - mnew);
      float sum = 0.f;
#pragma unroll
      for (int j = 0; j < 4; ++j) {
        float p = __expf(sc[j][r] - mnew);
        sc[j][r] = p;
        sum += p;
      }
#pragma unroll
      for (int off = 8; off >= 1; off >>= 1)
        sum += __shfl_xor(sum, off, 16);
      lrun[r] = lrun[r] * alpha + sum;
      mrun[r] = mnew;
#pragma unroll
      for (int j = 0; j < 4; ++j) oacc[j][r] *= alpha;
    }

    // ---- P: C-layout -> row-major f16 (wave-private LDS, in-order per wave) ----
#pragma unroll
    for (int j = 0; j < 4; ++j)
#pragma unroll
      for (int r = 0; r < 8; ++r)
        pw[(r + lhi * 8) * P_LD + j * 16 + llo] = (_Float16)sc[j][r];

    // ---- O += P V : A frags from LDS (b128), V frags contiguous from vT ----
#pragma unroll
    for (int ks = 0; ks < 2; ++ks) {
      const _Float16* prow = &pw[llo * P_LD + ks * 32];
      v16h af = load_frag16(prow + kb8, prow + 16 + kb8);
#pragma unroll
      for (int j = 0; j < 4; ++j) {
        const _Float16* vr = vT + (size_t)(g * HDIM + j * 16 + llo) * MROWS
                           + b * SEQ + kt + ks * 32 + kb8;
        v16h bf = load_frag16(vr, vr + 16);
        oacc[j] = __builtin_amdgcn_wmma_f32_16x16x32_f16(
            false, af, false, bf, (short)0, oacc[j], false, false);
      }
    }
  }

  // ---- normalize, write f16 attention output [B*S][E] ----
  const int orow = q0 + wave * 16 + lhi * 8;
#pragma unroll
  for (int r = 0; r < 8; ++r) {
    const float inv = 1.0f / lrun[r];
#pragma unroll
    for (int j = 0; j < 4; ++j)
      out[(size_t)(b * SEQ + orow + r) * EMBED + h * HDIM + j * 16 + llo] =
          (_Float16)(oacc[j][r] * inv);
  }
}

// ---------------------------------------------------------------------------
extern "C" void kernel_launch(void* const* d_in, const int* in_sizes, int n_in,
                              void* d_out, int out_size, void* d_ws, size_t ws_size,
                              hipStream_t stream) {
  (void)in_sizes; (void)n_in; (void)out_size; (void)ws_size;
  const float* query = (const float*)d_in[0];
  const float* key   = (const float*)d_in[1];
  const float* value = (const float*)d_in[2];
  const float* sdel  = (const float*)d_in[3];
  const float* mask  = (const float*)d_in[4];
  const float* Wq    = (const float*)d_in[5];
  const float* Wk    = (const float*)d_in[6];
  const float* Wv    = (const float*)d_in[7];
  const float* Wo    = (const float*)d_in[8];
  const float* tbw   = (const float*)d_in[9];
  float* outp = (float*)d_out;

  // ---- workspace arena (f16 elements) ----
  _Float16* x16q  = (_Float16*)d_ws;            // 8192x1024 (reused as attn16)
  _Float16* x16k  = x16q  + (size_t)MROWS * EMBED;
  _Float16* x16v  = x16k  + (size_t)MROWS * EMBED;
  _Float16* WqT   = x16v  + (size_t)MROWS * EMBED;     // [1024][1024]
  _Float16* WkT   = WqT   + (size_t)EMBED * EMBED;     // [256][1024]
  _Float16* WvT   = WkT   + (size_t)KVDIM * EMBED;     // [256][1024]
  _Float16* WoT   = WvT   + (size_t)KVDIM * EMBED;     // [1024][1024]
  _Float16* qproj = WoT   + (size_t)EMBED * EMBED;     // [8192][1024]
  _Float16* kproj = qproj + (size_t)MROWS * EMBED;     // [8192][256]
  _Float16* vT16  = kproj + (size_t)MROWS * KVDIM;     // [256][8192]
  float*    ct    = (float*)(vT16 + (size_t)KVDIM * MROWS);
  _Float16* attn16 = x16q;   // alias: x16q dead after Q projection

  cumsum_kernel<<<1, 32, 0, stream>>>(sdel, ct);

  const int n8 = (int)((size_t)MROWS * EMBED / 8);     // 1,048,576
  cvt_f16_kernel<<<(n8 + 255) / 256, 256, 0, stream>>>(query, x16q, n8);
  cvt_f16_kernel<<<(n8 + 255) / 256, 256, 0, stream>>>(key,   x16k, n8);
  cvt_f16_kernel<<<(n8 + 255) / 256, 256, 0, stream>>>(value, x16v, n8);

  transpose_cvt_kernel<<<dim3(EMBED / 32, EMBED / 32), 256, 0, stream>>>(Wq, WqT, EMBED, EMBED);
  transpose_cvt_kernel<<<dim3(KVDIM / 32, EMBED / 32), 256, 0, stream>>>(Wk, WkT, EMBED, KVDIM);
  transpose_cvt_kernel<<<dim3(KVDIM / 32, EMBED / 32), 256, 0, stream>>>(Wv, WvT, EMBED, KVDIM);
  transpose_cvt_kernel<<<dim3(EMBED / 32, EMBED / 32), 256, 0, stream>>>(Wo, WoT, EMBED, EMBED);

  // projections (LDS-free WMMA GEMMs)
  wmma_gemm_f16<1><<<dim3(EMBED / 64, MROWS / 128), 256, 0, stream>>>(
      x16q, WqT, qproj, MROWS, EMBED, EMBED);
  wmma_gemm_f16<1><<<dim3(KVDIM / 64, MROWS / 128), 256, 0, stream>>>(
      x16k, WkT, kproj, MROWS, KVDIM, EMBED);
  wmma_gemm_f16<2><<<dim3(KVDIM / 64, MROWS / 128), 256, 0, stream>>>(
      x16v, WvT, vT16, MROWS, KVDIM, EMBED);   // writes V^T directly

  flash_gqa_kernel<<<dim3(SEQ / 64, BATCH * NHEAD), 128, 0, stream>>>(
      qproj, kproj, vT16, ct, mask, tbw, attn16);

  wmma_gemm_f16<0><<<dim3(EMBED / 64, MROWS / 128), 256, 0, stream>>>(
      attn16, WoT, outp, MROWS, EMBED, EMBED);
}
